// LSTMEmbeddingModel_28905129902542
// MI455X (gfx1250) — matrix operand order repaired and checked by reading, not compiled, not run on hardware
//
#include <hip/hip_runtime.h>

// ---------------------------------------------------------------------------
// 2-layer bidirectional LSTM (T=32768, IN=1024, OUT=64, HID=40) + 2 heads.
// - Input projections: bf16 WMMA GEMMs (v_wmma_f32_16x16x32_bf16), with weight
//   fragments staged block-wide into LDS via GLOBAL_LOAD_ASYNC_TO_LDS_B128
//   (ASYNCcnt / s_wait_asynccnt) when the toolchain exposes the builtins.
// - Recurrences: LDS-resident sequential scans (latency-bound part).
// ---------------------------------------------------------------------------

#define T_LEN  32768
#define IN_DIM 1024
#define OUT_DIM 64
#define HID    40
#define XDIM   (IN_DIM + OUT_DIM)   // 1088
#define G4     (4 * HID)            // 160 gate rows per direction
#define NTOT   (2 * G4)             // 320 (both directions fused as N)
#define K1PAD  96                   // layer-1 K (2*HID=80) padded to 3*32
#define NTILES (NTOT / 16)          // 20 column tiles
#define KT0    (XDIM / 32)          // 34 k-tiles, layer 0
#define KT1    (K1PAD / 32)         // 3 k-tiles, layer 1

#define ASG __attribute__((address_space(1)))
#define ASL __attribute__((address_space(3)))

typedef __attribute__((ext_vector_type(16))) __bf16 v16bf;
typedef __attribute__((ext_vector_type(8)))  float  v8f;
typedef __attribute__((ext_vector_type(4)))  int    v4i;

union FragAB { v16bf v; unsigned int u32[8]; unsigned short u16[16]; };
union FragC  { v8f v; float f[8]; };

__device__ __forceinline__ unsigned short f2bf(float f) {
  unsigned int u = __builtin_bit_cast(unsigned int, f);
  u += 0x7FFFu + ((u >> 16) & 1u);          // round-to-nearest-even
  return (unsigned short)(u >> 16);
}
__device__ __forceinline__ float sigmoidf(float x) { return 1.0f / (1.0f + __expf(-x)); }

// ---------------------------------------------------------------------------
// Stage one N-column of WMMA B-fragments (kTiles x 1KB) into LDS, shared by
// all 8 waves of the block. Prefers the CDNA5 async global->LDS DMA path.
// Builtin signature (from round-2 diagnostic): arg0 = AS(1) v4i32* (global
// source), arg1 = AS(3) v4i32* (LDS dest), then imm offset, imm cpol.
// ---------------------------------------------------------------------------
#if defined(__has_builtin)
#  if __has_builtin(__builtin_amdgcn_global_load_async_to_lds_b128)
#    define HAVE_ASYNC_LDS 1
#  endif
#endif

__device__ __forceinline__ void stage_b(const unsigned short* __restrict__ wfrag,
                                        unsigned short* sB, int nt, int kTiles) {
  const int chunks = kTiles * 64;                     // 16-byte chunks per column
  for (int c = threadIdx.x; c < chunks; c += (int)blockDim.x) {
    const int kt = c >> 6;
    const int ch = c & 63;
    const size_t gOff = (size_t)(kt * NTILES + nt) * 1024 + (size_t)ch * 16;
    const int    lOff = kt * 1024 + ch * 16;
#ifdef HAVE_ASYNC_LDS
    __builtin_amdgcn_global_load_async_to_lds_b128(
        (ASG v4i*)(void*)((const char*)wfrag + gOff),
        (ASL v4i*)(void*)((char*)sB + lOff), 0, 0);
#else
    *(uint4*)((char*)sB + lOff) = *(const uint4*)((const char*)wfrag + gOff);
#endif
  }
#ifdef HAVE_ASYNC_LDS
#  if __has_builtin(__builtin_amdgcn_s_wait_asynccnt)
  __builtin_amdgcn_s_wait_asynccnt(0);
#  else
  asm volatile("s_wait_asynccnt 0x0" ::: "memory");
#  endif
#endif
  __syncthreads();
}

// ---------------------------------------------------------------------------
// Weight re-swizzle into per-wave WMMA B-fragments.
// B-fragment order used here (32x16 KxN, contiguous k): lane&15 = N column; lane>>4 selects
// K half (0..15 / 16..31); VGPR i holds K = half*16 + 2i, 2i+1.
// Storage: frag(kt,nt) -> 32 lanes x 16 bf16, contiguous 32 B per lane.
// ---------------------------------------------------------------------------
__global__ void k_prep_wfrag(const float* __restrict__ Wf, const float* __restrict__ Wb,
                             unsigned short* __restrict__ out, int kTiles, int kAct, int ldw) {
  int idx = blockIdx.x * blockDim.x + threadIdx.x;
  const int total = kTiles * NTILES * 512;
  if (idx >= total) return;
  const int e    = idx & 15;          // element within lane (VGPR i = e>>1, pair = e&1)
  const int lane = (idx >> 4) & 31;
  const int frag = idx >> 9;
  const int nt   = frag % NTILES;
  const int kt   = frag / NTILES;
  const int half = lane >> 4;
  const int n    = nt * 16 + (lane & 15);
  const int k    = kt * 32 + half * 16 + e;
  float v = 0.0f;
  if (k < kAct)
    v = (n < G4) ? Wf[(size_t)n * ldw + k] : Wb[(size_t)(n - G4) * ldw + k];
  out[idx] = f2bf(v);
}

__global__ void k_prep_bias(const float* a, const float* b, const float* c, const float* d,
                            float* __restrict__ bias) {
  int n = blockIdx.x * blockDim.x + threadIdx.x;
  if (n >= NTOT) return;
  bias[n] = (n < G4) ? (a[n] + b[n]) : (c[n - G4] + d[n - G4]);
}

__global__ void k_zero16(uint4* __restrict__ p, int n) {
  int i = blockIdx.x * blockDim.x + threadIdx.x;
  if (i < n) p[i] = make_uint4(0u, 0u, 0u, 0u);
}

// ---------------------------------------------------------------------------
// Layer-0 input projection: pre[t, 0:320] = concat(x,y)[t] @ [Wih_f;Wih_b]^T + b.
// A-fragments (16-bit 16x32 MxK): lane&15 = row M, lane>>4 selects K half;
// VGPR i: K = half*16 + 2i, 2i+1 (same contiguous k order as the B fragments).
// K tiles of 32 never straddle the x/y boundary (1024 % 32 == 0), so the
// source select is uniform per k-tile.
// ---------------------------------------------------------------------------
__global__ void k_pre0(const float* __restrict__ x, const float* __restrict__ y,
                       const unsigned short* __restrict__ wfrag,
                       const float* __restrict__ bias, float* __restrict__ pre) {
  __shared__ unsigned short sB[KT0 * 512];            // 34 KB of B fragments
  const int lane  = threadIdx.x & 31;
  const int wave  = threadIdx.x >> 5;
  const int half  = lane >> 4;
  const int l15   = lane & 15;
  const int mBase = blockIdx.x * 128 + wave * 16;
  const int nt    = blockIdx.y;
  const int nBase = nt * 16;
  const int rowA  = mBase + l15;

  stage_b(wfrag, sB, nt, KT0);

  const float* xrow = x + (size_t)rowA * IN_DIM;
  const float* yrow = y + (size_t)rowA * OUT_DIM;

  FragC acc;
#pragma unroll
  for (int r = 0; r < 8; ++r) acc.f[r] = 0.0f;

#pragma unroll 2
  for (int kt = 0; kt < KT0; ++kt) {
    const int kBase = kt * 32;
    const float* src = (kBase < IN_DIM) ? (xrow + kBase) : (yrow + (kBase - IN_DIM));
    FragAB a;
#pragma unroll
    for (int i = 0; i < 8; ++i) {
      const int kl = half * 16 + 2 * i;
      const float2 p = *(const float2*)(src + kl);
      a.u16[2 * i]     = f2bf(p.x);
      a.u16[2 * i + 1] = f2bf(p.y);
    }
    FragAB b;
    const uint4* bp = (const uint4*)((const char*)sB + ((size_t)(kt * 32 + lane) * 32));
    ((uint4*)b.u32)[0] = bp[0];                        // ds_load_b128
    ((uint4*)b.u32)[1] = bp[1];
    acc.v = __builtin_amdgcn_wmma_f32_16x16x32_bf16(false, a.v, false, b.v,
                                                    (short)0, acc.v, false, false);
  }
  const float bn = bias[nBase + l15];
#pragma unroll
  for (int r = 0; r < 8; ++r) {   // C/D layout: lanes 0-15 -> M=r, lanes 16-31 -> M=8+r
    const int row = mBase + half * 8 + r;
    pre[(size_t)row * NTOT + nBase + l15] = acc.f[r] + bn;
  }
}

// Layer-1 projection: A comes from the bf16 padded hidden buffer xcat1[T][96].
__global__ void k_pre1(const unsigned short* __restrict__ xcat1,
                       const unsigned short* __restrict__ wfrag,
                       const float* __restrict__ bias, float* __restrict__ pre) {
  __shared__ unsigned short sB[KT1 * 512];            // 3 KB of B fragments
  const int lane  = threadIdx.x & 31;
  const int wave  = threadIdx.x >> 5;
  const int half  = lane >> 4;
  const int l15   = lane & 15;
  const int mBase = blockIdx.x * 128 + wave * 16;
  const int nt    = blockIdx.y;
  const int nBase = nt * 16;
  const int rowA  = mBase + l15;

  stage_b(wfrag, sB, nt, KT1);

  const unsigned short* arow = xcat1 + (size_t)rowA * K1PAD;

  FragC acc;
#pragma unroll
  for (int r = 0; r < 8; ++r) acc.f[r] = 0.0f;

#pragma unroll
  for (int kt = 0; kt < KT1; ++kt) {
    const int kBase = kt * 32;
    FragAB a;
#pragma unroll
    for (int i = 0; i < 8; ++i) {
      const int kl = half * 16 + 2 * i;
      a.u32[i] = *(const unsigned int*)(arow + kBase + kl);
    }
    FragAB b;
    const uint4* bp = (const uint4*)((const char*)sB + ((size_t)(kt * 32 + lane) * 32));
    ((uint4*)b.u32)[0] = bp[0];
    ((uint4*)b.u32)[1] = bp[1];
    acc.v = __builtin_amdgcn_wmma_f32_16x16x32_bf16(false, a.v, false, b.v,
                                                    (short)0, acc.v, false, false);
  }
  const float bn = bias[nBase + l15];
#pragma unroll
  for (int r = 0; r < 8; ++r) {
    const int row = mBase + half * 8 + r;
    pre[(size_t)row * NTOT + nBase + l15] = acc.f[r] + bn;
  }
}

// ---------------------------------------------------------------------------
// Sequential LSTM scan. 2 blocks (dir 0 = fwd, dir 1 = bwd), 160 threads.
// Whh (160x40) + h/c/z live in LDS. Layer-0 emits bf16 hidden into xcat1;
// layer-1 emits only the final hidden per direction.
// ---------------------------------------------------------------------------
__global__ void k_scan(const float* __restrict__ pre,
                       const float* __restrict__ Whh_f, const float* __restrict__ Whh_b,
                       unsigned short* __restrict__ xcat1,  // layer0 output (or null)
                       float* __restrict__ hfinal) {        // layer1 output (or null)
  __shared__ float sW[G4 * HID];
  __shared__ float sh[HID];
  __shared__ float sc[HID];
  __shared__ float sz[G4];
  const int tid = threadIdx.x;     // 0..159, one gate row each
  const int dir = blockIdx.x;      // 0 fwd, 1 bwd
  const float* Whh = dir ? Whh_b : Whh_f;
  for (int i = tid; i < G4 * HID; i += G4) sW[i] = Whh[i];
  if (tid < HID) { sh[tid] = 0.0f; sc[tid] = 0.0f; }
  __syncthreads();

  for (int step = 0; step < T_LEN; ++step) {
    const int t = dir ? (T_LEN - 1 - step) : step;
    const float* prow = pre + (size_t)t * NTOT + dir * G4;
    if (tid == 0 && step + 1 < T_LEN) {
      const int tn = dir ? (t - 1) : (t + 1);
      __builtin_prefetch(pre + (size_t)tn * NTOT + dir * G4, 0, 1);  // global_prefetch
    }
    float acc = prow[tid];
    const float* wr = sW + tid * HID;
#pragma unroll 8
    for (int j = 0; j < HID; ++j) acc += wr[j] * sh[j];
    sz[tid] = acc;
    __syncthreads();
    if (tid < HID) {
      const float zi = sz[tid], zf = sz[HID + tid], zg = sz[2 * HID + tid], zo = sz[3 * HID + tid];
      const float cn = sigmoidf(zf) * sc[tid] + sigmoidf(zi) * tanhf(zg);
      const float hn = sigmoidf(zo) * tanhf(cn);
      sc[tid] = cn;
      sh[tid] = hn;
      if (xcat1) xcat1[(size_t)t * K1PAD + dir * HID + tid] = f2bf(hn);
    }
    __syncthreads();
  }
  if (hfinal && tid < HID) hfinal[dir * HID + tid] = sh[tid];
}

// ---------------------------------------------------------------------------
// Heads: emb[80] = [h2_fwd_last, h2_bwd_first]; e0 = Wh0@emb+bh0 (128),
// e1 = Wh1@emb+bh1 (64). d_out = [e0, e1] f32.
// ---------------------------------------------------------------------------
__global__ void k_heads(const float* __restrict__ h2,
                        const float* __restrict__ Wh0, const float* __restrict__ bh0,
                        const float* __restrict__ Wh1, const float* __restrict__ bh1,
                        float* __restrict__ out) {
  __shared__ float se[2 * HID];
  const int tid = threadIdx.x;
  if (tid < 2 * HID) se[tid] = h2[tid];
  __syncthreads();
  if (tid < 128) {
    float s = bh0[tid];
#pragma unroll 8
    for (int j = 0; j < 2 * HID; ++j) s += Wh0[tid * 80 + j] * se[j];
    out[tid] = s;
  } else if (tid < 192) {
    const int r = tid - 128;
    float s = bh1[r];
#pragma unroll 8
    for (int j = 0; j < 2 * HID; ++j) s += Wh1[r * 80 + j] * se[j];
    out[128 + r] = s;
  }
}

// ---------------------------------------------------------------------------
// Workspace layout (bytes, 128-aligned). Total ~46.7 MB (fits in 192 MB L2).
// ---------------------------------------------------------------------------
static constexpr size_t PRE_OFF   = 0;                                   // 32768*320*4
static constexpr size_t XCAT1_OFF = PRE_OFF + (size_t)T_LEN * NTOT * 4;  // 32768*96*2
static constexpr size_t WF0_OFF   = XCAT1_OFF + (size_t)T_LEN * K1PAD * 2;
static constexpr size_t WF1_OFF   = WF0_OFF + (size_t)KT0 * NTILES * 512 * 2;
static constexpr size_t B0_OFF    = WF1_OFF + (size_t)KT1 * NTILES * 512 * 2 + 128;
static constexpr size_t B1_OFF    = B0_OFF + NTOT * 4;
static constexpr size_t H2_OFF    = B1_OFF + NTOT * 4;

extern "C" void kernel_launch(void* const* d_in, const int* in_sizes, int n_in,
                              void* d_out, int out_size, void* d_ws, size_t ws_size,
                              hipStream_t stream) {
  const float* x        = (const float*)d_in[0];
  const float* y        = (const float*)d_in[1];
  const float* Wih_l0f  = (const float*)d_in[2];
  const float* Whh_l0f  = (const float*)d_in[3];
  const float* bih_l0f  = (const float*)d_in[4];
  const float* bhh_l0f  = (const float*)d_in[5];
  const float* Wih_l0b  = (const float*)d_in[6];
  const float* Whh_l0b  = (const float*)d_in[7];
  const float* bih_l0b  = (const float*)d_in[8];
  const float* bhh_l0b  = (const float*)d_in[9];
  const float* Wih_l1f  = (const float*)d_in[10];
  const float* Whh_l1f  = (const float*)d_in[11];
  const float* bih_l1f  = (const float*)d_in[12];
  const float* bhh_l1f  = (const float*)d_in[13];
  const float* Wih_l1b  = (const float*)d_in[14];
  const float* Whh_l1b  = (const float*)d_in[15];
  const float* bih_l1b  = (const float*)d_in[16];
  const float* bhh_l1b  = (const float*)d_in[17];
  const float* Wh0      = (const float*)d_in[18];
  const float* bh0      = (const float*)d_in[19];
  const float* Wh1      = (const float*)d_in[20];
  const float* bh1      = (const float*)d_in[21];

  char* ws = (char*)d_ws;
  float*          pre   = (float*)(ws + PRE_OFF);
  unsigned short* xcat1 = (unsigned short*)(ws + XCAT1_OFF);
  unsigned short* wf0   = (unsigned short*)(ws + WF0_OFF);
  unsigned short* wf1   = (unsigned short*)(ws + WF1_OFF);
  float*          b0    = (float*)(ws + B0_OFF);
  float*          b1    = (float*)(ws + B1_OFF);
  float*          h2    = (float*)(ws + H2_OFF);

  // --- prep: weight fragments, fused biases, zeroed padded layer-1 input ---
  k_prep_wfrag<<<(KT0 * NTILES * 512 + 255) / 256, 256, 0, stream>>>(
      Wih_l0f, Wih_l0b, wf0, KT0, XDIM, XDIM);
  k_prep_wfrag<<<(KT1 * NTILES * 512 + 255) / 256, 256, 0, stream>>>(
      Wih_l1f, Wih_l1b, wf1, KT1, 2 * HID, 2 * HID);
  k_prep_bias<<<2, 160, 0, stream>>>(bih_l0f, bhh_l0f, bih_l0b, bhh_l0b, b0);
  k_prep_bias<<<2, 160, 0, stream>>>(bih_l1f, bhh_l1f, bih_l1b, bhh_l1b, b1);
  const int nz = (int)((size_t)T_LEN * K1PAD * 2 / 16);
  k_zero16<<<(nz + 255) / 256, 256, 0, stream>>>((uint4*)xcat1, nz);

  // --- layer 0: WMMA projection, then sequential scan (f/b concurrent) ---
  k_pre0<<<dim3(T_LEN / 128, NTILES), 256, 0, stream>>>(x, y, wf0, b0, pre);
  k_scan<<<2, G4, 0, stream>>>(pre, Whh_l0f, Whh_l0b, xcat1, nullptr);

  // --- layer 1: WMMA projection (reusing pre), scan emitting final hiddens ---
  k_pre1<<<dim3(T_LEN / 128, NTILES), 256, 0, stream>>>(xcat1, wf1, b1, pre);
  k_scan<<<2, G4, 0, stream>>>(pre, Whh_l1f, Whh_l1b, nullptr, h2);

  // --- heads ---
  k_heads<<<1, 192, 0, stream>>>(h2, Wh0, bh0, Wh1, bh1, (float*)d_out);
}
